// GT_45561013076144
// MI455X (gfx1250) — matrix operand order repaired and checked by reading, not compiled, not run on hardware
//
#include <hip/hip_runtime.h>
#include <hip/hip_bf16.h>

// ---------------- problem constants (match reference) ----------------
#define NN     4096
#define NFEAT  512
#define NHID   256
#define NHEADS 4
#define NCLASS 64

// ---------------- WMMA types / helpers ----------------
typedef __attribute__((ext_vector_type(16))) __bf16        v16bf;
typedef __attribute__((ext_vector_type(8)))  float         v8f;
typedef __attribute__((ext_vector_type(8)))  unsigned short v8us;

union BFrag { v16bf v; unsigned short s[16]; };

__device__ __forceinline__ unsigned short f2bf(float f) {
  unsigned int u = __float_as_uint(f);
  unsigned int r = (u + 0x7FFFu + ((u >> 16) & 1u)) >> 16;          // RNE
  if ((u & 0x7F800000u) == 0x7F800000u && (u & 0x007FFFFFu))        // NaN: quiet it
    r = (u >> 16) | 0x40u;
  return (unsigned short)r;
}

__device__ __forceinline__ void load8(unsigned short* d, const unsigned short* s) {
  *(v8us*)d = *(const v8us*)s;   // 16B -> global_load_b128
}

__device__ __forceinline__ v8f wmma_bf16(v16bf a, v16bf b, v8f c) {
  // D = A(16x32 bf16) x B(32x16 bf16) + C(16x16 f32)
  return __builtin_amdgcn_wmma_f32_16x16x32_bf16(false, a, false, b, (short)0, c,
                                                 false, false);
}

// ---------------- f32 -> bf16 conversion ----------------
__global__ void cvt_bf16(const float* __restrict__ in, unsigned short* __restrict__ out,
                         int n) {
  int i = blockIdx.x * blockDim.x + threadIdx.x;
  if (i < n) out[i] = f2bf(in[i]);
}

// ---------------- GEMM: out[M x Nout] = A[M x K](bf16) @ W[Nout x K]^T(bf16) + bias
// one wave computes a 16x16 tile. outMode: 0=f32 row-major, 1=bf16 row-major,
// 2=bf16 transposed (out[n*ldo + m], used to produce V^T for attention).
__global__ void gemm_wmma(const unsigned short* __restrict__ A,
                          const unsigned short* __restrict__ W,
                          const float* __restrict__ bias,
                          void* __restrict__ outp,
                          int K, int ldo, int outMode) {
  const int lane = threadIdx.x;
  const int mt = blockIdx.x;
  const int nt = blockIdx.y * blockDim.y + threadIdx.y;
  const int r16 = lane & 15, hx = lane >> 4;
  const int m0 = mt * 16, n0 = nt * 16;

  v8f acc = {};
  // A frag: lane holds row m0+r16; slots e<8 -> k0+8*hx+e, e>=8 -> k0+16+8*hx+(e-8)
  const unsigned short* arow = A + (size_t)(m0 + r16) * K + hx * 8;
  // B frag: lane holds W row n0+r16 (column of B); slots e<8 -> k0+8*hx+e, e>=8 -> k0+16+8*hx+(e-8)
  const unsigned short* wrow = W + (size_t)(n0 + r16) * K + hx * 8;
  for (int k0 = 0; k0 < K; k0 += 32) {
    BFrag a, b;
    load8(&a.s[0], arow + k0);      load8(&a.s[8], arow + k0 + 16);
    load8(&b.s[0], wrow + k0);      load8(&b.s[8], wrow + k0 + 16);
    acc = wmma_bf16(a.v, b.v, acc);
  }
  const float bv = bias[n0 + r16];   // D layout: lane's 8 elems share column n0+r16
#pragma unroll
  for (int j = 0; j < 8; j++) {
    float val = acc[j] + bv;
    int m = m0 + j + 8 * hx;         // D: VGPR j -> row j + 8*half
    if (outMode == 0)
      ((float*)outp)[(size_t)m * ldo + n0 + r16] = val;
    else if (outMode == 1)
      ((unsigned short*)outp)[(size_t)m * ldo + n0 + r16] = f2bf(val);
    else
      ((unsigned short*)outp)[(size_t)(n0 + r16) * ldo + m] = f2bf(val);
  }
}

// ---------------- fused masked flash-attention ----------------
// One wave per (16-query tile, head). Computes S^T = K*Q^T so the C-layout puts a
// full query row's scores in lane pair (L, L^16); after online softmax the score
// fragments repack *in place* into the A-fragment layout for the P*V WMMA.
__global__ void attn_fused(const unsigned short* __restrict__ qb,  // [NN, H*C] bf16
                           const unsigned short* __restrict__ kb,  // [NN, H*C] bf16
                           const unsigned short* __restrict__ vT,  // [H*C, NN] bf16
                           const int* __restrict__ adj,            // [NN, NN]
                           float* __restrict__ om,                 // [NN, C*H] f32 (nch!)
                           int H, int C, float scale) {
  const int Nn = NN;
  const int HC = H * C;                 // C == 64 in all layers
  const int lane = threadIdx.x;
  const int hd = threadIdx.y;
  const int qt = blockIdx.x;
  const int r16 = lane & 15, hx = lane >> 4;
  const int nglob = qt * 16 + r16;      // query row owned by this lane

  // Q as B-fragments (two 32-wide c chunks): slots e<8 -> c = cc*32 + 8*hx + e, e>=8 -> c = cc*32 + 16 + 8*hx + (e-8)
  BFrag bq0, bq1;
  {
    const unsigned short* qp = qb + (size_t)nglob * HC + hd * C + hx * 8;
    load8(&bq0.s[0], qp);      load8(&bq0.s[8], qp + 16);
    load8(&bq1.s[0], qp + 32); load8(&bq1.s[8], qp + 48);
  }

  float mrun = -__builtin_inff();
  float lrun = 0.f;                     // partial row-sum (this lane's key halves)
  v8f o[4] = {};                        // 16x64 output tile (4 c-frames)
  const int* adjrow = adj + (size_t)nglob * Nn;

  for (int m0 = 0; m0 < Nn; m0 += 32) {
    // ---- S^T = K * Q^T for two 16-key tiles ----
    v8f s0 = {}, s1 = {};
    {
      BFrag ak;
      const unsigned short* kr = kb + (size_t)(m0 + r16) * HC + hd * C + hx * 8;
      load8(&ak.s[0], kr);      load8(&ak.s[8], kr + 16);
      s0 = wmma_bf16(ak.v, bq0.v, s0);
      load8(&ak.s[0], kr + 32); load8(&ak.s[8], kr + 48);
      s0 = wmma_bf16(ak.v, bq1.v, s0);
      const unsigned short* kr1 = kr + (size_t)16 * HC;
      load8(&ak.s[0], kr1);      load8(&ak.s[8], kr1 + 16);
      s1 = wmma_bf16(ak.v, bq0.v, s1);
      load8(&ak.s[0], kr1 + 32); load8(&ak.s[8], kr1 + 48);
      s1 = wmma_bf16(ak.v, bq1.v, s1);
    }
    // s0[j]: key m0 + 8*hx + j ; s1[j]: key m0 + 16 + 8*hx + j ; query = nglob

    // ---- mask with adj + online softmax ----
    float p[16];
    float cmax = -__builtin_inff();
    {
      const int* a0 = adjrow + m0 + hx * 8;
      const int* a1 = a0 + 16;
#pragma unroll
      for (int j = 0; j < 8; j++) {
        float v0 = a0[j] ? s0[j] * scale : -__builtin_inff();
        float v1 = a1[j] ? s1[j] * scale : -__builtin_inff();
        p[j] = v0; p[8 + j] = v1;
        cmax = fmaxf(cmax, fmaxf(v0, v1));
      }
    }
    cmax = fmaxf(cmax, __shfl_xor(cmax, 16, 32));   // combine key halves (lane pair)
    float mnew = fmaxf(mrun, cmax);
    float r = 1.f;
    if (mnew > -__builtin_inff()) {
      r = __expf(mrun - mnew);                      // exp(-inf)=0 handles first hit
      float lad = 0.f;
#pragma unroll
      for (int e = 0; e < 16; e++) { float pe = __expf(p[e] - mnew); p[e] = pe; lad += pe; }
      lrun = lrun * r + lad;
    } else {                                        // isolated row so far: alpha = 0
#pragma unroll
      for (int e = 0; e < 16; e++) p[e] = 0.f;
    }
    mrun = mnew;

    // ---- repack P (C-layout of S^T) directly into A-fragment layout ----
    // A slots e<8 -> k = 8*hx + e (== s0 frag), e>=8 -> k = 16 + 8*hx + (e-8) (== s1)
    BFrag pf;
#pragma unroll
    for (int e = 0; e < 16; e++) pf.s[e] = f2bf(p[e]);

    // ---- rescale accumulators: row of o[*][j] is (8*hx + j), r lives on that lane ----
#pragma unroll
    for (int j = 0; j < 8; j++) {
      float rj = __shfl(r, hx * 8 + j, 32);
      o[0][j] *= rj; o[1][j] *= rj; o[2][j] *= rj; o[3][j] *= rj;
    }

    // ---- o += P * V  (B frag from V^T: keys 8*hx+e, 16+8*hx+(e-8) per lane) ----
#pragma unroll
    for (int cf = 0; cf < 4; cf++) {
      BFrag bv;
      const unsigned short* vp =
          vT + (size_t)(hd * C + cf * 16 + r16) * Nn + m0 + hx * 8;
      load8(&bv.s[0], vp); load8(&bv.s[8], vp + 16);
      o[cf] = wmma_bf16(pf.v, bv.v, o[cf]);
    }
  }

  // ---- finalize: combine partial sums, normalize, store in nch (c-major/heads) ----
  lrun += __shfl_xor(lrun, 16, 32);
  float inv = (lrun > 0.f) ? 1.f / lrun : 0.f;      // isolated rows -> 0 (matches ref)
#pragma unroll
  for (int j = 0; j < 8; j++) {
    float ij = __shfl(inv, hx * 8 + j, 32);
    int node = qt * 16 + hx * 8 + j;
#pragma unroll
    for (int cf = 0; cf < 4; cf++) {
      int c = cf * 16 + r16;
      om[(size_t)node * HC + c * H + hd] = o[cf][j] * ij;  // col = c*H + h (nch quirk)
    }
  }
}

// ---------------- block reductions (blockDim.x <= 256, wave32) ----------------
__device__ __forceinline__ float block_sum(float v, float* sh) {
  int t = threadIdx.x;
#pragma unroll
  for (int ofs = 16; ofs > 0; ofs >>= 1) v += __shfl_xor(v, ofs, 32);
  __syncthreads();
  if ((t & 31) == 0) sh[t >> 5] = v;
  __syncthreads();
  float s = 0.f;
  int nw = (blockDim.x + 31) >> 5;
  for (int i = 0; i < nw; i++) s += sh[i];
  return s;
}
__device__ __forceinline__ float block_max(float v, float* sh) {
  int t = threadIdx.x;
#pragma unroll
  for (int ofs = 16; ofs > 0; ofs >>= 1) v = fmaxf(v, __shfl_xor(v, ofs, 32));
  __syncthreads();
  if ((t & 31) == 0) sh[t >> 5] = v;
  __syncthreads();
  float s = -__builtin_inff();
  int nw = (blockDim.x + 31) >> 5;
  for (int i = 0; i < nw; i++) s = fmaxf(s, sh[i]);
  return s;
}

// ---------------- beta gate + LayerNorm + ReLU (layers 0,1; one row per block) ----
__global__ void gate_ln_relu(const float* __restrict__ om, const float* __restrict__ xr,
                             const float* __restrict__ wb, const float* __restrict__ gam,
                             const float* __restrict__ bet, float* __restrict__ ho,
                             int D) {
  __shared__ float sh[8];
  int row = blockIdx.x, t = threadIdx.x;
  size_t idx = (size_t)row * D + t;
  float o = om[idx], x = xr[idx];
  float dot = block_sum(o * wb[t] + x * wb[D + t] + (o - x) * wb[2 * D + t], sh);
  float g = 1.f / (1.f + __expf(-dot));
  float hv = g * x + (1.f - g) * o;
  float s1 = block_sum(hv, sh);
  float s2 = block_sum(hv * hv, sh);
  float mu = s1 / (float)D;
  float var = s2 / (float)D - mu * mu;
  float y = (hv - mu) * rsqrtf(var + 1e-5f) * gam[t] + bet[t];
  ho[idx] = fmaxf(y, 0.f);
}

// ---------------- beta gate + log_softmax (layer 2) ----------------
__global__ void gate_logsoftmax(const float* __restrict__ om, const float* __restrict__ xr,
                                const float* __restrict__ wb, float* __restrict__ y,
                                int D) {
  __shared__ float sh[8];
  int row = blockIdx.x, t = threadIdx.x;
  size_t idx = (size_t)row * D + t;
  float o = om[idx], x = xr[idx];
  float dot = block_sum(o * wb[t] + x * wb[D + t] + (o - x) * wb[2 * D + t], sh);
  float g = 1.f / (1.f + __expf(-dot));
  float hv = g * x + (1.f - g) * o;
  float mx = block_max(hv, sh);
  float se = block_sum(__expf(hv - mx), sh);
  y[idx] = hv - mx - __logf(se);
}

// ---------------- orchestration ----------------
extern "C" void kernel_launch(void* const* d_in, const int* in_sizes, int n_in,
                              void* d_out, int out_size, void* d_ws, size_t ws_size,
                              hipStream_t stream) {
  (void)in_sizes; (void)n_in; (void)out_size; (void)ws_size;
  const float* x  = (const float*)d_in[0];
  const int* adj  = (const int*)d_in[1];
  // d_in flat order: x, adj, 3x{Wq,bq,Wk,bk,Wv,bv,Wskip,bskip,Wbeta}, 2x{gamma,beta}

  char* ws = (char*)d_ws;
  size_t off = 0;
  auto carve = [&](size_t bytes) -> void* {
    void* p = ws + off;
    off += (bytes + 255) & ~(size_t)255;
    return p;
  };
  unsigned short* act_bf = (unsigned short*)carve((size_t)NN * NFEAT * 2);
  unsigned short* wbf[4];
  for (int i = 0; i < 4; i++) wbf[i] = (unsigned short*)carve((size_t)NHID * NFEAT * 2);
  unsigned short* qbf = (unsigned short*)carve((size_t)NN * NHID * 2);
  unsigned short* kbf = (unsigned short*)carve((size_t)NN * NHID * 2);
  unsigned short* vtb = (unsigned short*)carve((size_t)NHID * NN * 2);   // V^T
  float* xr = (float*)carve((size_t)NN * NHID * 4);
  float* om = (float*)carve((size_t)NN * NHID * 4);
  float* hbuf0 = (float*)carve((size_t)NN * NHID * 4);
  float* hbuf1 = (float*)carve((size_t)NN * NHID * 4);

  const int fins[3] = {NFEAT, NHID, NHID};
  const int hcs[3]  = {NHID, NHID, NCLASS};
  const int hds[3]  = {NHEADS, NHEADS, 1};
  float* hbufs[2] = {hbuf0, hbuf1};

  const float* src = x;
  for (int l = 0; l < 3; l++) {
    const int fin = fins[l], hc = hcs[l], H = hds[l];
    void* const* P = d_in + 2 + l * 9;
    const float* Wq = (const float*)P[0];  const float* bq  = (const float*)P[1];
    const float* Wk = (const float*)P[2];  const float* bk  = (const float*)P[3];
    const float* Wv = (const float*)P[4];  const float* bv  = (const float*)P[5];
    const float* Ws = (const float*)P[6];  const float* bs  = (const float*)P[7];
    const float* Wbeta = (const float*)P[8];

    // convert activations + weights to bf16
    int na = NN * fin;
    cvt_bf16<<<dim3((na + 255) / 256), dim3(256), 0, stream>>>(src, act_bf, na);
    int nw = hc * fin;
    cvt_bf16<<<dim3((nw + 255) / 256), dim3(256), 0, stream>>>(Wq, wbf[0], nw);
    cvt_bf16<<<dim3((nw + 255) / 256), dim3(256), 0, stream>>>(Wk, wbf[1], nw);
    cvt_bf16<<<dim3((nw + 255) / 256), dim3(256), 0, stream>>>(Wv, wbf[2], nw);
    cvt_bf16<<<dim3((nw + 255) / 256), dim3(256), 0, stream>>>(Ws, wbf[3], nw);

    // projections: q,k bf16 row-major; v bf16 transposed; skip f32
    dim3 gg(NN / 16, hc / 64), gb(32, 4);
    gemm_wmma<<<gg, gb, 0, stream>>>(act_bf, wbf[0], bq, qbf, fin, hc, 1);
    gemm_wmma<<<gg, gb, 0, stream>>>(act_bf, wbf[1], bk, kbf, fin, hc, 1);
    gemm_wmma<<<gg, gb, 0, stream>>>(act_bf, wbf[2], bv, vtb, fin, NN, 2);
    gemm_wmma<<<gg, gb, 0, stream>>>(act_bf, wbf[3], bs, xr,  fin, hc, 0);

    // fused masked attention (scale = 1/sqrt(64))
    attn_fused<<<dim3(NN / 16), dim3(32, H), 0, stream>>>(qbf, kbf, vtb, adj, om,
                                                          H, 64, 0.125f);

    if (l < 2) {
      const float* gam = (const float*)d_in[29 + l * 2];
      const float* bet = (const float*)d_in[30 + l * 2];
      gate_ln_relu<<<dim3(NN), dim3(hc), 0, stream>>>(om, xr, Wbeta, gam, bet,
                                                      hbufs[l], hc);
      src = hbufs[l];
    } else {
      gate_logsoftmax<<<dim3(NN), dim3(hc), 0, stream>>>(om, xr, Wbeta,
                                                         (float*)d_out, hc);
    }
  }
}
